// Model_77412490543859
// MI455X (gfx1250) — matrix operand order repaired and checked by reading, not compiled, not run on hardware
//
#include <hip/hip_runtime.h>
#include <hip/hip_bf16.h>

// ---------------------------------------------------------------------------
// Types for CDNA5 WMMA
// ---------------------------------------------------------------------------
typedef __bf16 bf16_t;
typedef __attribute__((ext_vector_type(16))) __bf16 v16bf;
typedef __attribute__((ext_vector_type(8)))  float  v8f;

union U16F { v16bf v; unsigned u[8]; };

#define DEVINL static __device__ __forceinline__

DEVINL float relu6f(float x) { return fminf(fmaxf(x, 0.0f), 6.0f); }

DEVINL float wredsum(float v) {
  #pragma unroll
  for (int o = 16; o > 0; o >>= 1) v += __shfl_xor(v, o, 32);
  return v;
}

// ---------------------------------------------------------------------------
// Generic batched strided GEMM:  O = act(A @ W^T + bias)
//   A: element (m,k) at A[batch*sAb + m*sAm + k*sAk]   (f32)
//   W: (N,K) row-major, shared across batches           (f32)
//   O: element (m,n) at O[batch*sOb + m*sOm + n*sOn]    (f32)
// Block tile 128(M) x 128(N), 8 waves (wave32) arranged 4(M) x 2(N); each
// wave owns a 32x64 tile = 2x4 v_wmma_f32_16x16x32_bf16 accumulators.
// K stepped by 32 through LDS as bf16. Staging paths are split into a pure
// load phase (batched vmem, single wait) and a cvt+LDS-store phase.
// ---------------------------------------------------------------------------
__global__ void __launch_bounds__(256)
k_gemm(const float* __restrict__ A, const float* __restrict__ W,
       const float* __restrict__ bias, float* __restrict__ O,
       int M, int N, int K,
       long sAm, long sAk, long sAb,
       long sOm, long sOn, long sOb, int act)
{
  __shared__ bf16_t lA[128 * 32];
  __shared__ bf16_t lB[128 * 32];

  const int tid  = threadIdx.x;
  const int bM   = blockIdx.y * 128;
  const int bN   = blockIdx.x * 128;
  const float* Ab = A + (long)blockIdx.z * sAb;
  float*       Ob = O + (long)blockIdx.z * sOb;

  const int lane = tid & 31;
  const int wid  = tid >> 5;
  const int wM   = (wid & 3) * 32;   // 4 waves along M
  const int wN   = (wid >> 2) * 64;  // 2 waves along N; wave tile 32x64
  const int lr   = lane & 15;
  const int kh   = lane >> 4;

  v8f acc[2][4];
  #pragma unroll
  for (int a = 0; a < 2; a++)
    #pragma unroll
    for (int c = 0; c < 4; c++)
      #pragma unroll
      for (int i = 0; i < 8; i++) acc[a][c][i] = 0.0f;

  // Uniform (block-level) fast-path predicates.
  const bool fastA  = (sAk == 1) && ((sAm & 3) == 0) && (bM + 128 <= M);
  const bool fastAT = (sAm == 1) && ((sAk & 3) == 0) && (bM + 128 <= M);
  const bool fastW  = ((K & 3) == 0) && (bN + 128 <= N);

  for (int k0 = 0; k0 < K; k0 += 32) {
    const bool fullK = (k0 + 32 <= K);

    // ---- stage A tile (128x32) as bf16 ----
    if (fastA && fullK) {
      float4 f[4];
      #pragma unroll
      for (int i = 0; i < 4; i++) {
        int idx = i * 256 + tid;            // 1024 float4 chunks
        int m = idx >> 3, kq = (idx & 7) * 4;
        f[i] = *reinterpret_cast<const float4*>(
            Ab + (long)(bM + m) * sAm + (long)(k0 + kq));
      }
      #pragma unroll
      for (int i = 0; i < 4; i++) {
        int idx = i * 256 + tid;
        int m = idx >> 3, kq = (idx & 7) * 4;
        union { bf16_t h[4]; unsigned long long ull; } pk;
        pk.h[0] = (bf16_t)f[i].x; pk.h[1] = (bf16_t)f[i].y;
        pk.h[2] = (bf16_t)f[i].z; pk.h[3] = (bf16_t)f[i].w;
        *reinterpret_cast<unsigned long long*>(&lA[m * 32 + kq]) = pk.ull;
      }
    } else if (fastAT && fullK) {
      // m-contiguous A (transposed activations): float4 along m.
      float4 f[4];
      #pragma unroll
      for (int i = 0; i < 4; i++) {
        int idx = i * 256 + tid;
        int kk = idx >> 5, mq = (idx & 31) * 4;
        f[i] = *reinterpret_cast<const float4*>(
            Ab + (long)(bM + mq) + (long)(k0 + kk) * sAk);
      }
      #pragma unroll
      for (int i = 0; i < 4; i++) {
        int idx = i * 256 + tid;
        int kk = idx >> 5, mq = (idx & 31) * 4;
        lA[(mq + 0) * 32 + kk] = (bf16_t)f[i].x;
        lA[(mq + 1) * 32 + kk] = (bf16_t)f[i].y;
        lA[(mq + 2) * 32 + kk] = (bf16_t)f[i].z;
        lA[(mq + 3) * 32 + kk] = (bf16_t)f[i].w;
      }
    } else {
      float tmp[16];
      #pragma unroll
      for (int i = 0; i < 16; i++) {
        int idx = i * 256 + tid;
        int m = idx >> 5, kk = idx & 31;
        int mc = (bM + m) < M ? (bM + m) : (M - 1);
        int kc = (k0 + kk) < K ? (k0 + kk) : (K - 1);
        tmp[i] = Ab[(long)mc * sAm + (long)kc * sAk];
      }
      #pragma unroll
      for (int i = 0; i < 16; i++) {
        int idx = i * 256 + tid;
        int m = idx >> 5, kk = idx & 31;
        bool ok = (bM + m) < M && (k0 + kk) < K;
        lA[m * 32 + kk] = (bf16_t)(ok ? tmp[i] : 0.0f);
      }
    }

    // ---- stage W tile (128x32) as bf16 ----
    if (fastW && fullK) {
      float4 f[4];
      #pragma unroll
      for (int i = 0; i < 4; i++) {
        int idx = i * 256 + tid;
        int n = idx >> 3, kq = (idx & 7) * 4;
        f[i] = *reinterpret_cast<const float4*>(
            W + (long)(bN + n) * K + (long)(k0 + kq));
      }
      #pragma unroll
      for (int i = 0; i < 4; i++) {
        int idx = i * 256 + tid;
        int n = idx >> 3, kq = (idx & 7) * 4;
        union { bf16_t h[4]; unsigned long long ull; } pk;
        pk.h[0] = (bf16_t)f[i].x; pk.h[1] = (bf16_t)f[i].y;
        pk.h[2] = (bf16_t)f[i].z; pk.h[3] = (bf16_t)f[i].w;
        *reinterpret_cast<unsigned long long*>(&lB[n * 32 + kq]) = pk.ull;
      }
    } else {
      float tmp[16];
      #pragma unroll
      for (int i = 0; i < 16; i++) {
        int idx = i * 256 + tid;
        int n = idx >> 5, kk = idx & 31;
        int nc = (bN + n) < N ? (bN + n) : (N - 1);
        int kc = (k0 + kk) < K ? (k0 + kk) : (K - 1);
        tmp[i] = W[(long)nc * K + (long)kc];
      }
      #pragma unroll
      for (int i = 0; i < 16; i++) {
        int idx = i * 256 + tid;
        int n = idx >> 5, kk = idx & 31;
        bool ok = (bN + n) < N && (k0 + kk) < K;
        lB[n * 32 + kk] = (bf16_t)(ok ? tmp[i] : 0.0f);
      }
    }
    __syncthreads();

    // prefetch next K tile (global_prefetch_b8)
    if (k0 + 32 < K) {
      int pm = bM + (tid >> 1);
      int pk = k0 + 32 + (tid & 1) * 16;
      if (pm < M && pk < K)
        __builtin_prefetch(&Ab[(long)pm * sAm + (long)pk * sAk], 0, 3);
      int pn = bN + (tid & 127);
      if (pn < N)
        __builtin_prefetch(&W[(long)pn * K + (k0 + 32)], 0, 3);
    }

    // ---- B fragments (32x16 each; K = (i>=4?16:0) + kh*8 + 2(i&3) per ISA layout) ----
    U16F bfr[4];
    #pragma unroll
    for (int nt = 0; nt < 4; nt++)
      #pragma unroll
      for (int i = 0; i < 8; i++) {
        int kk = ((i & 4) ? 16 : 0) + kh * 8 + (i & 3) * 2;
        bfr[nt].u[i] = *(const unsigned*)&lB[(wN + nt * 16 + lr) * 32 + kk];
      }
    // ---- A fragments + 8 WMMA ----
    #pragma unroll
    for (int mt = 0; mt < 2; mt++) {
      U16F af;
      int m = wM + mt * 16 + lr;
      #pragma unroll
      for (int i = 0; i < 8; i++) {
        int kk = ((i & 4) ? 16 : 0) + kh * 8 + (i & 3) * 2;
        af.u[i] = *(const unsigned*)&lA[m * 32 + kk];
      }
      #pragma unroll
      for (int nt = 0; nt < 4; nt++)
        acc[mt][nt] = __builtin_amdgcn_wmma_f32_16x16x32_bf16(
            false, af.v, false, bfr[nt].v, (short)0, acc[mt][nt], false, false);
    }
    __syncthreads();
  }

  // ---- epilogue: bias + optional relu6, strided store ----
  if (bM + 128 <= M && bN + 128 <= N) {
    #pragma unroll
    for (int mt = 0; mt < 2; mt++)
      #pragma unroll
      for (int nt = 0; nt < 4; nt++)
        #pragma unroll
        for (int i = 0; i < 8; i++) {
          int row = bM + wM + mt * 16 + kh * 8 + i;
          int col = bN + wN + nt * 16 + lr;
          float v = acc[mt][nt][i];
          if (bias) v += bias[col];
          if (act)  v = relu6f(v);
          Ob[(long)row * sOm + (long)col * sOn] = v;
        }
  } else {
    #pragma unroll
    for (int mt = 0; mt < 2; mt++)
      #pragma unroll
      for (int nt = 0; nt < 4; nt++)
        #pragma unroll
        for (int i = 0; i < 8; i++) {
          int row = bM + wM + mt * 16 + kh * 8 + i;
          int col = bN + wN + nt * 16 + lr;
          if (row < M && col < N) {
            float v = acc[mt][nt][i];
            if (bias) v += bias[col];
            if (act)  v = relu6f(v);
            Ob[(long)row * sOm + (long)col * sOn] = v;
          }
        }
  }
}

// ---------------------------------------------------------------------------
// LayerNorm (one wave per row), optional residual input and relu6.
// ---------------------------------------------------------------------------
template <int LL>
__global__ void __launch_bounds__(256)
k_layernorm(const float* __restrict__ x, const float* __restrict__ res,
            const float* __restrict__ g, const float* __restrict__ bta,
            float* __restrict__ out, int rows, int act)
{
  int gw   = (blockIdx.x * blockDim.x + threadIdx.x) >> 5;
  int lane = threadIdx.x & 31;
  if (gw >= rows) return;
  const int NC = LL / 32;
  const float* xr = x + (long)gw * LL;
  const float* rr = res ? res + (long)gw * LL : nullptr;
  float vals[NC];
  float s = 0.0f;
  #pragma unroll
  for (int i = 0; i < NC; i++) {
    float v = xr[lane + 32 * i];
    if (rr) v += rr[lane + 32 * i];
    vals[i] = v; s += v;
  }
  s = wredsum(s);
  float mean = s / (float)LL;
  float var = 0.0f;
  #pragma unroll
  for (int i = 0; i < NC; i++) { float d = vals[i] - mean; var += d * d; }
  var = wredsum(var) / (float)LL;
  float inv = rsqrtf(var + 1e-5f);
  float* orow = out + (long)gw * LL;
  #pragma unroll
  for (int i = 0; i < NC; i++) {
    float y = (vals[i] - mean) * inv * g[lane + 32 * i] + bta[lane + 32 * i];
    if (act) y = relu6f(y);
    orow[lane + 32 * i] = y;
  }
}

// ---------------------------------------------------------------------------
// loc = (xyrot @ loc1^T + b1) @ loc2^T + b2     (B,256)
// ---------------------------------------------------------------------------
__global__ void k_loc(const float* __restrict__ xyrot,
                      const float* __restrict__ w1, const float* __restrict__ b1,
                      const float* __restrict__ w2, const float* __restrict__ b2,
                      float* __restrict__ loc)
{
  int idx = blockIdx.x * 256 + threadIdx.x;
  if (idx >= 128 * 256) return;
  int n = idx & 255, b = idx >> 8;
  float x0 = xyrot[b * 4 + 0], x1 = xyrot[b * 4 + 1];
  float x2 = xyrot[b * 4 + 2], x3 = xyrot[b * 4 + 3];
  float acc = b2[n];
  #pragma unroll
  for (int j = 0; j < 16; j++) {
    float h = b1[j] + x0 * w1[j * 4 + 0] + x1 * w1[j * 4 + 1]
                    + x2 * w1[j * 4 + 2] + x3 * w1[j * 4 + 3];
    acc += w2[n * 16 + j] * h;
  }
  loc[idx] = acc;
}

// ---------------------------------------------------------------------------
// Gather piece embeddings pp (B,14,256); current = pp[:,0] + loc
// ---------------------------------------------------------------------------
__global__ void k_pieces(const int* __restrict__ queue, const int* __restrict__ remains,
                         const float* __restrict__ emb, const float* __restrict__ loc,
                         float* __restrict__ pp, float* __restrict__ cur)
{
  int idx = blockIdx.x * 256 + threadIdx.x;
  if (idx >= 128 * 14 * 256) return;
  int c = idx & 255;
  int t = (idx >> 8) % 14;
  int b = idx / (14 * 256);
  int piece = (t < 7) ? queue[b * 7 + t] : remains[b * 7 + (t - 7)];
  float e = emb[piece * 256 + c];
  pp[idx] = e;
  if (t == 0) cur[b * 256 + c] = e + loc[b * 256 + c];
}

// ---------------------------------------------------------------------------
// Conv front: dw1 (1ch 3x3 VALID 22x12->20x10) + pw1 (1->128) + bn1 + relu6.
// Output t1 in (B,200,128) layout (spatial-major for the pw2 GEMM).
// ---------------------------------------------------------------------------
__global__ void __launch_bounds__(256)
k_conv_front(const float* __restrict__ board,
             const float* __restrict__ dw1w, const float* __restrict__ dw1b,
             const float* __restrict__ pw1w, const float* __restrict__ pw1b,
             const float* __restrict__ g, const float* __restrict__ bta,
             const float* __restrict__ mu, const float* __restrict__ vr,
             float* __restrict__ t1)
{
  __shared__ float dval[200];
  int b = blockIdx.x;
  const float* bd = board + (long)b * 22 * 12;
  for (int s = threadIdx.x; s < 200; s += 256) {
    int y = s / 10, x = s % 10;
    float acc = dw1b[0];
    #pragma unroll
    for (int dy = 0; dy < 3; dy++)
      #pragma unroll
      for (int dx = 0; dx < 3; dx++)
        acc += bd[(y + dy) * 12 + (x + dx)] * dw1w[dy * 3 + dx];
    dval[s] = acc;
  }
  __syncthreads();
  for (int idx = threadIdx.x; idx < 200 * 128; idx += 256) {
    int s = idx >> 7, c = idx & 127;
    float v = dval[s] * pw1w[c] + pw1b[c];
    float sc = g[c] * rsqrtf(vr[c] + 1e-5f);
    v = (v - mu[c]) * sc + bta[c];
    t1[((long)b * 200 + s) * 128 + c] = relu6f(v);
  }
}

// ---------------------------------------------------------------------------
// Depthwise 3x3 SAME conv on (B,200,128) spatial-major tensor.
// ---------------------------------------------------------------------------
__global__ void k_dw2(const float* __restrict__ t1, const float* __restrict__ w,
                      const float* __restrict__ bb, float* __restrict__ t2)
{
  long idx = (long)blockIdx.x * 256 + threadIdx.x;
  if (idx >= (long)128 * 200 * 128) return;
  int c = idx & 127;
  long r = idx >> 7;
  int s = (int)(r % 200);
  int b = (int)(r / 200);
  int y = s / 10, x = s % 10;
  float acc = bb[c];
  #pragma unroll
  for (int dy = 0; dy < 3; dy++) {
    int yy = y + dy - 1;
    if (yy < 0 || yy >= 20) continue;
    #pragma unroll
    for (int dx = 0; dx < 3; dx++) {
      int xx = x + dx - 1;
      if (xx < 0 || xx >= 10) continue;
      acc += t1[((long)b * 200 + yy * 10 + xx) * 128 + c] * w[c * 9 + dy * 3 + dx];
    }
  }
  t2[idx] = acc;
}

// ---------------------------------------------------------------------------
// bn2 + relu6 + 2D positional encoding; produces state (B,200,256).
// ---------------------------------------------------------------------------
__global__ void k_bn2pos(const float* __restrict__ in,
                         const float* __restrict__ g, const float* __restrict__ bta,
                         const float* __restrict__ mu, const float* __restrict__ vr,
                         float* __restrict__ state)
{
  long idx = (long)blockIdx.x * 256 + threadIdx.x;
  if (idx >= (long)25600 * 256) return;
  int c = idx & 255;
  long row = idx >> 8;
  int s = (int)(row % 200);
  int y = s / 10, x = s % 10;
  float v = in[idx];
  float sc = g[c] * rsqrtf(vr[c] + 1e-5f);
  v = relu6f((v - mu[c]) * sc + bta[c]);
  int j = c >> 1;
  float div = powf(10000.0f, (float)(4 * j) / 256.0f);
  float pe = (c & 1) ? cosf((float)x / div) : sinf((float)y / div);
  state[idx] = v + pe;
}

// ---------------------------------------------------------------------------
// Pieces attention: one wave per (b, qpos, head); lane = head dim (DH=32).
// s_k = q.k_k/sqrt(32) + bias[b,n,q,k]; softmax over k; o = sum w_k v_k.
// ---------------------------------------------------------------------------
template <int KLEN>
__global__ void __launch_bounds__(256)
k_attn_pieces(const float* __restrict__ q, const float* __restrict__ k,
              const float* __restrict__ v, const float* __restrict__ bias,
              float* __restrict__ o)
{
  int gw   = (blockIdx.x * blockDim.x + threadIdx.x) >> 5;
  int lane = threadIdx.x & 31;
  if (gw >= 128 * 200 * 8) return;
  int n = gw & 7, qp = (gw >> 3) % 200, b = gw / (8 * 200);
  float qd = q[((long)(b * 200 + qp)) * 256 + n * 32 + lane];
  const float* kp = k + ((long)b * KLEN) * 256 + n * 32 + lane;
  const float* vp = v + ((long)b * KLEN) * 256 + n * 32 + lane;
  const float* bp = bias + (((long)(b * 8 + n)) * 200 + qp) * KLEN;
  float s[KLEN];
  float mx = -1e30f;
  #pragma unroll
  for (int j = 0; j < KLEN; j++) {
    float p = wredsum(qd * kp[(long)j * 256]);
    p = p * 0.17677669529663687f + bp[j];
    s[j] = p; mx = fmaxf(mx, p);
  }
  float den = 0.0f;
  #pragma unroll
  for (int j = 0; j < KLEN; j++) { s[j] = expf(s[j] - mx); den += s[j]; }
  float acc = 0.0f;
  #pragma unroll
  for (int j = 0; j < KLEN; j++) acc += s[j] * vp[(long)j * 256];
  o[((long)(b * 200 + qp)) * 256 + n * 32 + lane] = acc / den;
}

// ---------------------------------------------------------------------------
// Final attention: 1 query per batch over 200 keys, online softmax.
// One wave per (b, head).
// ---------------------------------------------------------------------------
__global__ void __launch_bounds__(256)
k_attn_final(const float* __restrict__ q, const float* __restrict__ k,
             const float* __restrict__ v, float* __restrict__ o)
{
  int gw   = (blockIdx.x * blockDim.x + threadIdx.x) >> 5;
  int lane = threadIdx.x & 31;
  if (gw >= 128 * 8) return;
  int n = gw & 7, b = gw >> 3;
  float qd = q[b * 256 + n * 32 + lane];
  const float* kp = k + ((long)b * 200) * 256 + n * 32 + lane;
  const float* vp = v + ((long)b * 200) * 256 + n * 32 + lane;
  float m = -1e30f, den = 0.0f, acc = 0.0f;
  for (int j = 0; j < 200; j++) {
    float s = wredsum(qd * kp[(long)j * 256]) * 0.17677669529663687f;
    float nm = fmaxf(m, s);
    float cf = expf(m - nm);
    float w  = expf(s - nm);
    den = den * cf + w;
    acc = acc * cf + w * vp[(long)j * 256];
    m = nm;
  }
  o[b * 256 + n * 32 + lane] = acc / den;
}

// ---------------------------------------------------------------------------
// Host orchestration
// ---------------------------------------------------------------------------
static inline void gemm(hipStream_t st, const float* A, const float* W,
                        const float* bias, float* O,
                        int M, int N, int K,
                        long sAm, long sAk, long sAb,
                        long sOm, long sOn, long sOb, int nb, int act)
{
  dim3 grid((N + 127) / 128, (M + 127) / 128, nb);
  k_gemm<<<grid, 256, 0, st>>>(A, W, bias, O, M, N, K,
                               sAm, sAk, sAb, sOm, sOn, sOb, act);
}

extern "C" void kernel_launch(void* const* d_in, const int* in_sizes, int n_in,
                              void* d_out, int out_size, void* d_ws, size_t ws_size,
                              hipStream_t stream)
{
  (void)in_sizes; (void)n_in; (void)out_size; (void)ws_size;
  const float* xyrot   = (const float*)d_in[0];
  const int*   queue   = (const int*)d_in[1];
  const int*   remains = (const int*)d_in[2];
  const float* board   = (const float*)d_in[3];
  // d_in[4] = garbage (unused by reference)
  const float* const* P = (const float* const*)(d_in + 5);

  enum {
    LOC1_W = 0, LOC1_B, LOC2_W, LOC2_B, PIECE_EMB, ROT_W, ROT_B,
    Q_W, Q_B, K_W, K_B, V_W, V_B, QKV_W, QKV_B, QL_W, QL_B,
    AN_G, AN_B, FN_G, FN_B,
    DW1_W, DW1_B, PW1_W, PW1_B, BN1_G, BN1_B, BN1_M, BN1_V,
    DW2_W, DW2_B, PW2_W, PW2_B, BN2_G, BN2_B, BN2_M, BN2_V,
    MHA_IN_W, MHA_IN_B, MHA_OUT_W, MHA_OUT_B, POL_W, POL_B, VAL_W, VAL_B,
    AB_L1_W, AB_L2_W, AB_L2_B, AB_LN1_G, AB_LN1_B, AB_L3_W, AB_L3_B,
    AB_LN2_G, AB_LN2_B, AB_L4_W, AB_L4_B,
    AL_L1_W, AL_L2_W, AL_L2_B, AL_LN1_G, AL_LN1_B, AL_L3_W, AL_L3_B,
    AL_LN2_G, AL_LN2_B, AL_L4_W, AL_L4_B
  };

  float* ws = (float*)d_ws;
  long off = 0;
  auto alloc = [&](long n) { float* p = ws + off; off += n; return p; };
  float* state = alloc(6553600);   // (B,200,256)
  float* newb  = alloc(6553600);   // residual / pw2 pre-bn
  float* qb    = alloc(6553600);   // q proj; also t1; also final K
  float* ob    = alloc(6553600);   // attn out; also t2; also final V
  float* pr    = alloc(1835008);   // (B,14,4,256) rotations
  float* pp    = alloc(458752);    // (B,14,256)
  float* loc   = alloc(32768);
  float* cur   = alloc(32768);
  float* h1    = alloc(1048576);   // (B,8192)
  float* h2    = alloc(16384);     // (B,128)
  float* h3    = alloc(131072);    // (B,1024)
  float* hb    = alloc(5734400);   // bias tab (B,8,200,olen<=28)
  float* kb    = alloc(917504);    // (B,28,256)
  float* vb    = alloc(917504);
  float* qc    = alloc(32768);     // final query (B,256)
  float* ofin  = alloc(32768);
  float* xb    = alloc(32768);

  float* logits = (float*)d_out;          // (B,10)
  float* value  = (float*)d_out + 1280;   // (B,256)

  // --- front: loc, pieces, rotations ---
  k_loc<<<128, 256, 0, stream>>>(xyrot, P[LOC1_W], P[LOC1_B], P[LOC2_W], P[LOC2_B], loc);
  k_pieces<<<(128 * 14 * 256) / 256, 256, 0, stream>>>(queue, remains, P[PIECE_EMB], loc, pp, cur);
  gemm(stream, pp, P[ROT_W], P[ROT_B], pr, 1792, 1024, 256, 256, 1, 0, 1024, 1, 0, 1, 0);

  // --- conv stack -> state ---
  k_conv_front<<<128, 256, 0, stream>>>(board, P[DW1_W], P[DW1_B], P[PW1_W], P[PW1_B],
                                        P[BN1_G], P[BN1_B], P[BN1_M], P[BN1_V], qb);
  k_dw2<<<(128 * 200 * 128 + 255) / 256, 256, 0, stream>>>(qb, P[DW2_W], P[DW2_B], ob);
  gemm(stream, ob, P[PW2_W], P[PW2_B], newb, 25600, 256, 128, 128, 1, 0, 256, 1, 0, 1, 0);
  k_bn2pos<<<25600, 256, 0, stream>>>(newb, P[BN2_G], P[BN2_B], P[BN2_M], P[BN2_V], state);

  auto bias_layer = [&](int l1w, int l2w, int l2b, int g1, int b1,
                        int l3w, int l3b, int g2, int b2, int l4w, int l4b, int ol) {
    // (a) state^T @ l1^T per batch -> h1 in (B, 32*256) with j = k*256 + c; relu6
    gemm(stream, state, P[l1w], nullptr, h1, 256, 32, 200,
         1, 256, 51200, 1, 256, 8192, 128, 1);
    // (b) h1 @ l2^T + b, then LN + relu6
    gemm(stream, h1, P[l2w], P[l2b], h2, 128, 128, 8192, 8192, 1, 0, 128, 1, 0, 1, 0);
    k_layernorm<128><<<16, 256, 0, stream>>>(h2, nullptr, P[g1], P[b1], h2, 128, 1);
    // (c) h2 @ l3^T + b -> (B,8,128), LN + relu6 per (b,nh)
    gemm(stream, h2, P[l3w], P[l3b], h3, 128, 1024, 128, 128, 1, 0, 1024, 1, 0, 1, 0);
    k_layernorm<128><<<128, 256, 0, stream>>>(h3, nullptr, P[g2], P[b2], h3, 1024, 1);
    // (d) (B*8,128) @ l4^T + b -> bias tab (B,8,200,ol)  [attn reads transposed]
    gemm(stream, h3, P[l4w], P[l4b], hb, 1024, 200 * ol, 128,
         128, 1, 0, (long)200 * ol, 1, 0, 1, 0);
  };

  auto attn_block = [&](const float* piecesA, int klen) {
    gemm(stream, state, P[Q_W], P[Q_B], qb, 25600, 256, 256, 256, 1, 0, 256, 1, 0, 1, 0);
    gemm(stream, piecesA, P[K_W], P[K_B], kb, klen, 256, 256,
         256, 1, 14336, 256, 1, (long)klen * 256, 128, 0);
    gemm(stream, piecesA, P[V_W], P[V_B], vb, klen, 256, 256,
         256, 1, 14336, 256, 1, (long)klen * 256, 128, 0);
    if (klen == 4)
      k_attn_pieces<4><<<25600, 256, 0, stream>>>(qb, kb, vb, hb, ob);
    else
      k_attn_pieces<28><<<25600, 256, 0, stream>>>(qb, kb, vb, hb, ob);
    gemm(stream, ob, P[QKV_W], P[QKV_B], newb, 25600, 256, 256, 256, 1, 0, 256, 1, 0, 1, 0);
    k_layernorm<256><<<3200, 256, 0, stream>>>(state, newb, P[AN_G], P[AN_B], state, 25600, 0);
    gemm(stream, state, P[QL_W], P[QL_B], newb, 25600, 256, 256, 256, 1, 0, 256, 1, 0, 1, 0);
    k_layernorm<256><<<3200, 256, 0, stream>>>(state, newb, P[FN_G], P[FN_B], state, 25600, 0);
  };

  // --- 7 visible-piece iterations ---
  for (int i = 0; i < 7; i++) {
    bias_layer(AB_L1_W, AB_L2_W, AB_L2_B, AB_LN1_G, AB_LN1_B,
               AB_L3_W, AB_L3_B, AB_LN2_G, AB_LN2_B, AB_L4_W, AB_L4_B, 4);
    attn_block(pr + (long)i * 1024, 4);
  }
  // --- final not-visible pass (klen = 28) ---
  bias_layer(AL_L1_W, AL_L2_W, AL_L2_B, AL_LN1_G, AL_LN1_B,
             AL_L3_W, AL_L3_B, AL_LN2_G, AL_LN2_B, AL_L4_W, AL_L4_B, 28);
  attn_block(pr + 7 * 1024, 28);

  // --- final single-query MHA ---
  gemm(stream, cur, P[MHA_IN_W], P[MHA_IN_B], qc, 128, 256, 256,
       256, 1, 0, 256, 1, 0, 1, 0);
  gemm(stream, state, P[MHA_IN_W] + 256 * 256, P[MHA_IN_B] + 256, qb,
       25600, 256, 256, 256, 1, 0, 256, 1, 0, 1, 0);
  gemm(stream, state, P[MHA_IN_W] + 512 * 256, P[MHA_IN_B] + 512, ob,
       25600, 256, 256, 256, 1, 0, 256, 1, 0, 1, 0);
  k_attn_final<<<32, 256, 0, stream>>>(qc, qb, ob, ofin);

  gemm(stream, ofin, P[MHA_OUT_W], P[MHA_OUT_B], xb, 128, 256, 256,
       256, 1, 0, 256, 1, 0, 1, 0);
  gemm(stream, xb, P[POL_W], P[POL_B], logits, 128, 10, 256,
       256, 1, 0, 10, 1, 0, 1, 0);
  gemm(stream, xb, P[VAL_W], P[VAL_B], value, 128, 256, 256,
       256, 1, 0, 256, 1, 0, 1, 0);
}
